// SimpleAlignmentModel_15556371546616
// MI455X (gfx1250) — matrix operand order repaired and checked by reading, not compiled, not run on hardware
//
// SimpleAlignmentModel forward for MI455X (gfx1250), wave32 + WMMA bf16.
// GEMM core: v_wmma_f32_16x16x32_bf16, one 16x64 strip per wave (4 accumulators,
// A-fragment reuse), 8 waves/block. All call sites have contiguous K, K%32==0,
// M%16==0, N%64==0, so the kernel carries no per-element bounds checks at all.
// fp32 -> bf16 operand conversion uses native __bf16 casts (hardware cvt ops).
#include <hip/hip_runtime.h>
#include <math.h>

#define BB 4
#define NN 512
#define TT 2048
#define DEC_C 512
#define SP_C 768
#define HH 512
#define BC 64
#define NH 8
#define HD 8
#define D_INNER 1024
#define D_STATE 16
#define DT_RANK 32
#define NEGV (-1000000000.0f)
#define LOG2PI 1.8378770664093453f

typedef __attribute__((ext_vector_type(16))) __bf16 v16bf;
typedef __attribute__((ext_vector_type(8)))  float  v8f;

union BFVec { __bf16 h[16]; v16bf v; };
union F16U  { float4 q[4]; float f[16]; };

__device__ inline float siluf(float x) { return x / (1.f + __expf(-x)); }
__device__ inline float act_f(float x, int act) {
  if (act == 1) return siluf(x);
  if (act == 2) return x > 0.f ? x : 0.f;
  if (act == 3) return x > 20.f ? x : log1pf(__expf(x));   // softplus
  return x;
}

// ---------------------------------------------------------------------------
// WMMA GEMM: C[m,n] (+)= act( sum_k A[m,k]*B[n,k] + bias[n] )
// A[z*bsA + m*sAm + k], B[z*bsB + n*sBn + k], C[z*bsC + m*sCm + n].
// Requirements (true for every call site): K%32==0, M%16==0, N%64==0,
// sAm%4==0, sBn%4==0, contiguous K.
// Wave computes 16 rows x 64 cols; block = 8 waves = 128 rows x 64 cols.
// ---------------------------------------------------------------------------
__global__ __launch_bounds__(256)
void k_wmma_gemm(const float* __restrict__ A, const float* __restrict__ Bw,
                 const float* __restrict__ bias, float* __restrict__ C,
                 int M, int N, int K,
                 long sAm, long sBn, long sCm,
                 long bsA, long bsB, long bsC, int act, int accum)
{
  const int lane = threadIdx.x & 31;
  const int wave = threadIdx.x >> 5;
  const int m0 = blockIdx.y * 128 + wave * 16;
  const int n0 = blockIdx.x * 64;
  if (m0 >= M) return;
  A  += (long)blockIdx.z * bsA;
  Bw += (long)blockIdx.z * bsB;
  C  += (long)blockIdx.z * bsC;
  const int r    = lane & 15;
  const int half = lane >> 4;
  // A fragment (contiguous k order, same as B): elems 0-15 -> K = k0+16*half+e
  const float* Arow = A + (long)(m0 + r) * sAm + half * 16;
  // B fragment: elems 0-15 -> K = k0+16*half+e, row = n-col
  const float* B0 = Bw + (long)(n0 + r) * sBn + half * 16;
  const float* B1 = B0 + 16 * sBn;
  const float* B2 = B1 + 16 * sBn;
  const float* B3 = B2 + 16 * sBn;

  v8f acc[4];
  acc[0] = {}; acc[1] = {}; acc[2] = {}; acc[3] = {};

  for (int k0 = 0; k0 < K; k0 += 32) {
    BFVec a;
    {
      F16U t;
      const float4* p0 = (const float4*)(Arow + k0);
      const float4* p1 = (const float4*)(Arow + k0 + 8);
      t.q[0] = p0[0]; t.q[1] = p0[1]; t.q[2] = p1[0]; t.q[3] = p1[1];
      #pragma unroll
      for (int e = 0; e < 16; ++e) a.h[e] = (__bf16)t.f[e];
    }
    const float* Bp[4] = {B0, B1, B2, B3};
    #pragma unroll
    for (int j = 0; j < 4; ++j) {
      F16U t;
      const float4* p = (const float4*)(Bp[j] + k0);
      t.q[0] = p[0]; t.q[1] = p[1]; t.q[2] = p[2]; t.q[3] = p[3];
      BFVec b;
      #pragma unroll
      for (int e = 0; e < 16; ++e) b.h[e] = (__bf16)t.f[e];
      acc[j] = __builtin_amdgcn_wmma_f32_16x16x32_bf16(false, a.v, false, b.v,
                                                       (short)0, acc[j], false, false);
    }
  }

  // C/D layout: elem e -> M = m0 + e + 8*half, N = n0 + j*16 + r
  #pragma unroll
  for (int j = 0; j < 4; ++j) {
    const int col = n0 + j * 16 + r;
    const float bv = bias ? bias[col] : 0.f;
    #pragma unroll
    for (int e = 0; e < 8; ++e) {
      const int cm = m0 + e + half * 8;
      float v = act_f(acc[j][e] + bv, act);
      long idx = (long)cm * sCm + col;
      if (accum) v += C[idx];
      C[idx] = v;
    }
  }
}

// ---------------------------------------------------------------------------
// Elementwise / norm / conv / attention / scan / DP kernels
// ---------------------------------------------------------------------------
__global__ void k_transpose(const float* __restrict__ in, float* __restrict__ out,
                            int L, int C) {  // (B,L,C) -> (B,C,L)
  long i = blockIdx.x * (long)blockDim.x + threadIdx.x;
  long total = (long)BB * L * C;
  if (i >= total) return;
  int c = (int)(i % C); long t = i / C; int l = (int)(t % L); int b = (int)(t / L);
  out[((long)b * C + c) * L + l] = in[i];
}

__global__ void k_conv_enc(const float* __restrict__ x, const float* __restrict__ w,
                           const float* __restrict__ bias, float* __restrict__ out,
                           int C, int L) {  // out = silu(x + conv5_same(x) + b), (B,C,L)
  long i = blockIdx.x * (long)blockDim.x + threadIdx.x;
  long total = (long)BB * C * L;
  if (i >= total) return;
  int l = (int)(i % L); long t = i / L; int c = (int)(t % C); int b = (int)(t / C);
  const float* xb = x + (long)b * C * L;
  const float* wr = w + (long)c * C * 5;
  float accv = bias[c] + x[i];
  for (int ci = 0; ci < C; ++ci) {
    const float* xr = xb + (long)ci * L;
    #pragma unroll
    for (int kk = 0; kk < 5; ++kk) {
      int ll = l + kk - 2;
      if (ll >= 0 && ll < L) accv += xr[ll] * wr[ci * 5 + kk];
    }
  }
  out[i] = siluf(accv);
}

__global__ void k_gn_stats(const float* __restrict__ x, float* __restrict__ st,
                           int C, int L, int G) {  // block per (b,g)
  __shared__ float sh[256], sh2[256];
  int b = blockIdx.x / G, g = blockIdx.x % G;
  long n = (long)(C / G) * L;
  const float* base = x + ((long)b * C + (long)g * (C / G)) * L;
  float s = 0.f, s2 = 0.f;
  for (long i = threadIdx.x; i < n; i += 256) { float v = base[i]; s += v; s2 += v * v; }
  sh[threadIdx.x] = s; sh2[threadIdx.x] = s2; __syncthreads();
  for (int t = 128; t > 0; t >>= 1) {
    if (threadIdx.x < t) { sh[threadIdx.x] += sh[threadIdx.x + t]; sh2[threadIdx.x] += sh2[threadIdx.x + t]; }
    __syncthreads();
  }
  if (threadIdx.x == 0) {
    float m = sh[0] / n, var = sh2[0] / n - m * m;
    st[blockIdx.x * 2] = m; st[blockIdx.x * 2 + 1] = rsqrtf(var + 1e-5f);
  }
}

__global__ void k_gn_apply(const float* __restrict__ x, const float* __restrict__ st,
                           const float* __restrict__ gam, const float* __restrict__ bet,
                           float* __restrict__ out, int C, int L, int G, int transposeOut) {
  long i = blockIdx.x * (long)blockDim.x + threadIdx.x;
  long total = (long)BB * C * L;
  if (i >= total) return;
  int l = (int)(i % L); long t = i / L; int c = (int)(t % C); int b = (int)(t / C);
  int g = c / (C / G);
  float m = st[(b * G + g) * 2], inv = st[(b * G + g) * 2 + 1];
  float v = (x[i] - m) * inv * gam[c] + bet[c];
  if (transposeOut) out[((long)b * L + l) * C + c] = v; else out[i] = v;
}

__global__ void k_ln(const float* __restrict__ x, const float* __restrict__ g,
                     const float* __restrict__ bsrc, float* __restrict__ out,
                     int D, int rms) {  // block per row
  __shared__ float sh[256];
  const float* row = x + (long)blockIdx.x * D;
  float* orow = out + (long)blockIdx.x * D;
  float s = 0.f, s2 = 0.f;
  for (int i = threadIdx.x; i < D; i += 256) { float v = row[i]; s += v; s2 += v * v; }
  sh[threadIdx.x] = s; __syncthreads();
  for (int t = 128; t > 0; t >>= 1) { if (threadIdx.x < t) sh[threadIdx.x] += sh[threadIdx.x + t]; __syncthreads(); }
  s = sh[0]; __syncthreads();
  sh[threadIdx.x] = s2; __syncthreads();
  for (int t = 128; t > 0; t >>= 1) { if (threadIdx.x < t) sh[threadIdx.x] += sh[threadIdx.x + t]; __syncthreads(); }
  s2 = sh[0]; __syncthreads();
  float m = rms ? 0.f : s / D;
  float inv = rsqrtf(s2 / D - m * m + 1e-5f);
  for (int i = threadIdx.x; i < D; i += 256)
    orow[i] = (row[i] - m) * inv * g[i] + (bsrc ? bsrc[i] : 0.f);
}

__global__ void k_scores(const float* __restrict__ Q, const float* __restrict__ Kh,
                         float* __restrict__ S) {
  long i = blockIdx.x * (long)blockDim.x + threadIdx.x;
  long total = (long)BB * NH * NN * TT;
  if (i >= total) return;
  int kk = (int)(i % TT); long t = i / TT;
  int q = (int)(t % NN); t /= NN;
  int h = (int)(t % NH); int b = (int)(t / NH);
  const float* qp = Q + ((long)b * NN + q) * BC + h * HD;
  const float* kp = Kh + ((long)b * TT + kk) * BC + h * HD;
  float s = 0.f;
  #pragma unroll
  for (int d = 0; d < HD; ++d) s += qp[d] * kp[d];
  S[i] = s * 0.3535533906f;  // 1/sqrt(8)
}

__global__ void k_softmax(float* __restrict__ x, int D) {  // block per row
  __shared__ float sh[256];
  float* row = x + (long)blockIdx.x * D;
  float mx = -1e30f;
  for (int i = threadIdx.x; i < D; i += 256) mx = fmaxf(mx, row[i]);
  sh[threadIdx.x] = mx; __syncthreads();
  for (int t = 128; t > 0; t >>= 1) { if (threadIdx.x < t) sh[threadIdx.x] = fmaxf(sh[threadIdx.x], sh[threadIdx.x + t]); __syncthreads(); }
  mx = sh[0]; __syncthreads();
  float sum = 0.f;
  for (int i = threadIdx.x; i < D; i += 256) { float e = __expf(row[i] - mx); row[i] = e; sum += e; }
  sh[threadIdx.x] = sum; __syncthreads();
  for (int t = 128; t > 0; t >>= 1) { if (threadIdx.x < t) sh[threadIdx.x] += sh[threadIdx.x + t]; __syncthreads(); }
  float inv = 1.f / sh[0];
  __syncthreads();
  for (int i = threadIdx.x; i < D; i += 256) row[i] *= inv;
}

__global__ void k_mha_o(const float* __restrict__ S, const float* __restrict__ V,
                        float* __restrict__ O) {
  long i = blockIdx.x * (long)blockDim.x + threadIdx.x;
  long total = (long)BB * NN * BC;
  if (i >= total) return;
  int hd = (int)(i % BC); long t = i / BC; int q = (int)(t % NN); int b = (int)(t / NN);
  int h = hd / HD;
  const float* srow = S + (((long)b * NH + h) * NN + q) * TT;
  const float* vcol = V + (long)b * TT * BC + hd;
  float s = 0.f;
  for (int kk = 0; kk < TT; ++kk) s += srow[kk] * vcol[(long)kk * BC];
  O[i] = s;
}

__global__ void k_rT_xmr(const float* __restrict__ sT, float* __restrict__ rT,
                         float* __restrict__ xmr) {
  long i = blockIdx.x * (long)blockDim.x + threadIdx.x;
  long total = (long)BB * NN * HH;
  if (i >= total) return;
  int c = (int)(i % HH); long t = i / HH; int n = (int)(t % NN); int b = (int)(t / NN);
  float xm = sT[((long)b * 2 * HH + c) * NN + n];
  float xl = sT[((long)b * 2 * HH + HH + c) * NN + n];
  float rr = __expf(-2.f * xl);
  rT[i] = rr; xmr[i] = xm * rr;
}

__global__ void k_mzc2(const float* __restrict__ zp, float* __restrict__ out) {
  long i = blockIdx.x * (long)blockDim.x + threadIdx.x;
  long total = (long)BB * TT * HH;
  if (i >= total) return;
  float v = zp[i];
  out[i] = -0.5f * v * v;
}

__global__ void k_l14(const float* __restrict__ sT, const float* __restrict__ rT,
                      float* __restrict__ l14) {
  int i = blockIdx.x * blockDim.x + threadIdx.x;
  if (i >= BB * NN) return;
  int n = i % NN, b = i / NN;
  const float* xm_base = sT + (long)b * 2 * HH * NN + n;
  const float* xl_base = xm_base + (long)HH * NN;
  const float* rrow = rT + ((long)b * NN + n) * HH;
  float s = 0.f;
  for (int c = 0; c < HH; ++c) {
    float xm = xm_base[(long)c * NN];
    float xl = xl_base[(long)c * NN];
    s += -0.5f * LOG2PI - xl - 0.5f * xm * xm * rrow[c];
  }
  l14[i] = s;
}

__global__ void k_addl14(float* __restrict__ logp, const float* __restrict__ l14) {
  long i = blockIdx.x * (long)blockDim.x + threadIdx.x;
  long total = (long)BB * NN * TT;
  if (i >= total) return;
  logp[i] += l14[i / TT];
}

__global__ void k_mp_forward(const float* __restrict__ logp, const int* __restrict__ xl,
                             const int* __restrict__ zl, float* __restrict__ Q) {
  __shared__ float prev[NN];
  int b = blockIdx.x, x = threadIdx.x;
  int xlen = xl[b], zlen = zl[b];
  prev[x] = NEGV; __syncthreads();
  for (int y = 0; y < TT; ++y) {
    float vy = (x < xlen && y < zlen) ? logp[((long)b * NN + x) * TT + y] : 0.f;
    float p = prev[x];
    float pm1 = (x > 0) ? prev[x - 1] : NEGV;
    __syncthreads();
    float v_cur = (x == y) ? NEGV : p;
    float v_prev = (x == 0) ? ((y == 0) ? 0.f : NEGV) : pm1;
    int hi = (xlen - 1 < y) ? (xlen - 1) : y;
    bool band = (x >= xlen + y - zlen) && (x <= hi);
    float cur = band ? (fmaxf(v_cur, v_prev) + vy) : vy;
    prev[x] = cur;
    Q[((long)b * TT + y) * NN + x] = cur;
    __syncthreads();
  }
}

__global__ void k_mp_backtrack(const float* __restrict__ Q, const int* __restrict__ xl,
                               const int* __restrict__ zl, int* __restrict__ idxout) {
  int b = threadIdx.x;
  if (b >= BB) return;
  int idx = xl[b] - 1;
  int zlen = zl[b];
  for (int y = TT - 1; y >= 0; --y) {
    int ym1 = (y - 1 > 0) ? (y - 1) : 0;
    const float* Qm1 = Q + ((long)b * TT + ym1) * NN;
    bool valid = y < zlen;
    idxout[b * TT + y] = valid ? idx : -1;
    float q_stay = Qm1[idx];
    float q_move = Qm1[(idx - 1 > 0) ? (idx - 1) : 0];
    bool mv = (idx != 0) && ((idx == y) || (q_move > q_stay));
    if (valid && y > 0 && mv) idx--;
  }
}

__global__ void k_gather(const float* __restrict__ sT, const int* __restrict__ idx,
                         float* __restrict__ am, float* __restrict__ al) {
  long i = blockIdx.x * (long)blockDim.x + threadIdx.x;
  long total = (long)BB * HH * TT;
  if (i >= total) return;
  int t = (int)(i % TT); long u = i / TT; int c = (int)(u % HH); int b = (int)(u / HH);
  int id = idx[b * TT + t];
  float vm = 0.f, vl = 0.f;
  if (id >= 0) {
    vm = sT[((long)b * 2 * HH + c) * NN + id];
    vl = sT[((long)b * 2 * HH + HH + c) * NN + id];
  }
  am[i] = vm; al[i] = vl;
}

__global__ void k_sampled(const float* __restrict__ am, const float* __restrict__ al,
                          const float* __restrict__ noise, const int* __restrict__ zl,
                          float* __restrict__ out) {
  long i = blockIdx.x * (long)blockDim.x + threadIdx.x;
  long total = (long)BB * TT * HH;
  if (i >= total) return;
  int c = (int)(i % HH); long u = i / HH; int t = (int)(u % TT); int b = (int)(u / TT);
  long j = ((long)b * HH + c) * TT + t;
  float zm = (t < zl[b]) ? 1.f : 0.f;
  out[i] = am[j] + __expf(al[j]) * noise[j] * zm;
}

__global__ void k_conv_dw(const float* __restrict__ xz, const float* __restrict__ w,
                          const float* __restrict__ bias, float* __restrict__ out, int L) {
  long i = blockIdx.x * (long)blockDim.x + threadIdx.x;
  long total = (long)BB * L * D_INNER;
  if (i >= total) return;
  int d = (int)(i % D_INNER); long t = i / D_INNER; int l = (int)(t % L); int b = (int)(t / L);
  float a = bias[d];
  #pragma unroll
  for (int kk = 0; kk < 4; ++kk) {
    int ll = l - 3 + kk;
    if (ll >= 0) a += xz[((long)b * L + ll) * (2 * D_INNER) + d] * w[d * 4 + kk];
  }
  out[i] = siluf(a);
}

__global__ void k_scan(const float* __restrict__ delta, const float* __restrict__ dbc,
                       const float* __restrict__ xi, const float* __restrict__ Alog,
                       const float* __restrict__ Dp, float* __restrict__ y, int L) {
  int tid = blockIdx.x * blockDim.x + threadIdx.x;
  if (tid >= BB * D_INNER) return;
  int b = tid / D_INNER, d = tid % D_INNER;
  float A[D_STATE], h[D_STATE];
  #pragma unroll
  for (int s = 0; s < D_STATE; ++s) { A[s] = -__expf(Alog[d * D_STATE + s]); h[s] = 0.f; }
  float Dd = Dp[d];
  for (int l = 0; l < L; ++l) {
    long rl = (long)b * L + l;
    float dt = delta[rl * D_INNER + d];
    float xv = xi[rl * D_INNER + d];
    const float* bm = dbc + rl * 64 + DT_RANK;
    const float* cm = bm + D_STATE;
    float acc = 0.f;
    #pragma unroll
    for (int s = 0; s < D_STATE; ++s) {
      h[s] = __expf(dt * A[s]) * h[s] + dt * bm[s] * xv;
      acc += h[s] * cm[s];
    }
    y[rl * D_INNER + d] = acc + Dd * xv;
  }
}

__global__ void k_resmul(float* __restrict__ y, const float* __restrict__ xz, int L) {
  long i = blockIdx.x * (long)blockDim.x + threadIdx.x;
  long total = (long)BB * L * D_INNER;
  if (i >= total) return;
  int d = (int)(i % D_INNER); long bl = i / D_INNER;
  y[i] *= siluf(xz[bl * (2 * D_INNER) + D_INNER + d]);
}

__global__ void k_add(const float* __restrict__ a, const float* __restrict__ b,
                      float* __restrict__ o, long n) {
  long i = blockIdx.x * (long)blockDim.x + threadIdx.x;
  if (i >= n) return;
  o[i] = a[i] + b[i];
}

// ---------------------------------------------------------------------------
// Host side
// ---------------------------------------------------------------------------
static inline dim3 g1(long n) { return dim3((unsigned)((n + 255) / 256)); }

static void gemm_full(hipStream_t st, const float* A, const float* Bw, const float* bias,
                      float* C, int M, int N, int K,
                      long sAm, long sBn, long sCm,
                      long bsA, long bsB, long bsC, int nb, int act, int accum) {
  dim3 grid(N / 64, (M + 127) / 128, nb);
  k_wmma_gemm<<<grid, 256, 0, st>>>(A, Bw, bias, C, M, N, K,
                                    sAm, sBn, sCm, bsA, bsB, bsC, act, accum);
}
static void gemm_rm(hipStream_t st, const float* A, const float* Bw, const float* bias,
                    float* C, int M, int N, int K, int act, int accum) {
  gemm_full(st, A, Bw, bias, C, M, N, K, K, K, N, 0, 0, 0, 1, act, accum);
}

struct EncP { const float *ffw, *ffb, *cw, *cb, *gg, *gb; };
static void run_encoder(hipStream_t st, const float* in, int L, int Cin, int Cout,
                        EncP p, float* out, float* t0, float* t1, float* t2, float* stats) {
  gemm_rm(st, in, p.ffw, p.ffb, t0, BB * L, Cout, Cin, 1 /*silu*/, 0);
  long n1 = (long)BB * L * Cout;
  k_transpose<<<g1(n1), 256, 0, st>>>(t0, t1, L, Cout);
  k_conv_enc<<<g1(n1), 256, 0, st>>>(t1, p.cw, p.cb, t2, Cout, L);
  k_gn_stats<<<BB * 8, 256, 0, st>>>(t2, stats, Cout, L, 8);
  k_gn_apply<<<g1(n1), 256, 0, st>>>(t2, stats, p.gg, p.gb, out, Cout, L, 8, 1);
}

// input index map (setup_inputs dict order, params flattened in insertion order)
enum {
  IN_X = 0, IN_XLENS, IN_Z, IN_ZLENS, IN_NOISE,
  P_INDEC_W, P_INDEC_B, P_INSP_W, P_INSP_B,
  P_LAT = 9, P_BNDEC = 15, P_BNSP = 21, P_QP = 27, P_KP = 33, P_VP = 39,
  P_MHA_INW = 45, P_MHA_INB, P_MHA_OUTW, P_MHA_OUTB,
  P_APW = 49, P_APB,
  P_ST1W = 51, P_ST1B, P_ST2W, P_ST2B, P_ST3W, P_ST3B,
  P_ANG = 57, P_ANB,
  P_LN1G = 59, P_LN1B, P_LN2G, P_LN2B,
  P_LAYER0 = 63, P_LAYER1 = 73,   // per layer: rms_w,in_w,conv_w,conv_b,xproj_w,dt_w,dt_b,A_log,D,out_w
  P_FFN1W = 83, P_FFN1B, P_FFN2W, P_FFN2B, P_FPW, P_FPB
};

// workspace layout (floats). Phase-1 peak ~75.6M floats (~303 MB);
// mamba arena time-overlaps the then-dead phase-1 region.
static const long F_ZP = 0;
static const long F_X    = F_ZP    + (long)BB * TT * HH;
static const long F_T0   = F_X     + (long)BB * NN * DEC_C;
static const long F_T1   = F_T0    + (long)BB * TT * HH;
static const long F_T2   = F_T1    + (long)BB * TT * HH;
static const long F_STATS= F_T2    + (long)BB * TT * HH;
static const long F_BX   = F_STATS + 256;
static const long F_BZ   = F_BX    + (long)BB * NN * BC;
static const long F_Q    = F_BZ    + (long)BB * TT * BC;
static const long F_K    = F_Q     + (long)BB * NN * BC;
static const long F_V    = F_K     + (long)BB * TT * BC;
static const long F_QH   = F_V     + (long)BB * TT * BC;
static const long F_KH   = F_QH    + (long)BB * NN * BC;
static const long F_VH   = F_KH    + (long)BB * TT * BC;
static const long F_SC   = F_VH    + (long)BB * TT * BC;
static const long F_AO   = F_SC    + (long)BB * NH * NN * TT;
static const long F_AOUT = F_AO    + (long)BB * NN * BC;
static const long F_S1   = F_AOUT  + (long)BB * NN * BC;
static const long F_S2   = F_S1    + (long)BB * NN * HH;
static const long F_S    = F_S2    + (long)BB * NN * HH;
static const long F_ST   = F_S     + (long)BB * NN * 2 * HH;
static const long F_RT   = F_ST    + (long)BB * NN * 2 * HH;
static const long F_XMR  = F_RT    + (long)BB * NN * HH;
static const long F_MZ   = F_XMR   + (long)BB * NN * HH;
static const long F_LOGP = F_MZ    + (long)BB * TT * HH;
static const long F_L14  = F_LOGP  + (long)BB * NN * TT;
static const long F_QDP  = F_L14   + (long)BB * NN;
static const long F_IDX  = F_QDP   + (long)BB * TT * NN;
// mamba arena (time-overlapped with phase-1 buffers, all dead by then)
static const long F_SAMP = 0;
static const long F_HBUF = F_SAMP  + (long)BB * TT * HH;
static const long F_MTMP = F_HBUF  + (long)BB * TT * HH;
static const long F_XZ   = F_MTMP  + (long)BB * TT * HH;
static const long F_XCONV= F_XZ    + (long)BB * TT * 2 * D_INNER;
static const long F_DBC  = F_XCONV + (long)BB * TT * D_INNER;
static const long F_DELTA= F_DBC   + (long)BB * TT * 64;
static const long F_YS   = F_DELTA + (long)BB * TT * D_INNER;
static const long F_X2   = F_YS    + (long)BB * TT * D_INNER;
static const long F_FFH  = F_X2    + (long)BB * TT * HH;

static const long OUT_AM   = 0;
static const long OUT_AL   = OUT_AM   + (long)BB * HH * TT;
static const long OUT_LAT  = OUT_AL   + (long)BB * HH * TT;
static const long OUT_TRUE = OUT_LAT  + (long)BB * TT * HH;
static const long OUT_PRED = OUT_TRUE + (long)BB * TT * SP_C;

extern "C" void kernel_launch(void* const* d_in, const int* in_sizes, int n_in,
                              void* d_out, int out_size, void* d_ws, size_t ws_size,
                              hipStream_t stream) {
  (void)in_sizes; (void)n_in; (void)out_size; (void)ws_size;
  auto F = [&](int i) { return (const float*)d_in[i]; };
  const int* xlens = (const int*)d_in[IN_XLENS];
  const int* zlens = (const int*)d_in[IN_ZLENS];
  float* W = (float*)d_ws;
  float* out = (float*)d_out;
  float* outAM = out + OUT_AM;
  float* outAL = out + OUT_AL;
  float* outLAT = out + OUT_LAT;

  // x = linear(x, in_dec)
  gemm_rm(stream, F(IN_X), F(P_INDEC_W), F(P_INDEC_B), W + F_X, BB * NN, HH, DEC_C, 0, 0);

  // latent_feats = encoder(z) -> straight into d_out
  EncP lat{F(P_LAT), F(P_LAT + 1), F(P_LAT + 2), F(P_LAT + 3), F(P_LAT + 4), F(P_LAT + 5)};
  run_encoder(stream, F(IN_Z), TT, SP_C, HH, lat, outLAT, W + F_T0, W + F_T1, W + F_T2, W + F_STATS);

  // zp = linear(z, in_sp)
  gemm_rm(stream, F(IN_Z), F(P_INSP_W), F(P_INSP_B), W + F_ZP, BB * TT, HH, SP_C, 0, 0);

  EncP bd{F(P_BNDEC), F(P_BNDEC + 1), F(P_BNDEC + 2), F(P_BNDEC + 3), F(P_BNDEC + 4), F(P_BNDEC + 5)};
  run_encoder(stream, W + F_X, NN, HH, BC, bd, W + F_BX, W + F_T0, W + F_T1, W + F_T2, W + F_STATS);
  EncP bs{F(P_BNSP), F(P_BNSP + 1), F(P_BNSP + 2), F(P_BNSP + 3), F(P_BNSP + 4), F(P_BNSP + 5)};
  run_encoder(stream, W + F_ZP, TT, HH, BC, bs, W + F_BZ, W + F_T0, W + F_T1, W + F_T2, W + F_STATS);
  EncP qp{F(P_QP), F(P_QP + 1), F(P_QP + 2), F(P_QP + 3), F(P_QP + 4), F(P_QP + 5)};
  run_encoder(stream, W + F_BX, NN, BC, BC, qp, W + F_Q, W + F_T0, W + F_T1, W + F_T2, W + F_STATS);
  EncP kp{F(P_KP), F(P_KP + 1), F(P_KP + 2), F(P_KP + 3), F(P_KP + 4), F(P_KP + 5)};
  run_encoder(stream, W + F_BZ, TT, BC, BC, kp, W + F_K, W + F_T0, W + F_T1, W + F_T2, W + F_STATS);
  EncP vp{F(P_VP), F(P_VP + 1), F(P_VP + 2), F(P_VP + 3), F(P_VP + 4), F(P_VP + 5)};
  run_encoder(stream, W + F_BZ, TT, BC, BC, vp, W + F_V, W + F_T0, W + F_T1, W + F_T2, W + F_STATS);

  // MHA
  gemm_rm(stream, W + F_Q, F(P_MHA_INW),               F(P_MHA_INB),          W + F_QH, BB * NN, BC, BC, 0, 0);
  gemm_rm(stream, W + F_K, F(P_MHA_INW) + BC * BC,     F(P_MHA_INB) + BC,     W + F_KH, BB * TT, BC, BC, 0, 0);
  gemm_rm(stream, W + F_V, F(P_MHA_INW) + 2 * BC * BC, F(P_MHA_INB) + 2 * BC, W + F_VH, BB * TT, BC, BC, 0, 0);
  k_scores<<<g1((long)BB * NH * NN * TT), 256, 0, stream>>>(W + F_QH, W + F_KH, W + F_SC);
  k_softmax<<<BB * NH * NN, 256, 0, stream>>>(W + F_SC, TT);
  k_mha_o<<<g1((long)BB * NN * BC), 256, 0, stream>>>(W + F_SC, W + F_VH, W + F_AO);
  gemm_rm(stream, W + F_AO, F(P_MHA_OUTW), F(P_MHA_OUTB), W + F_AOUT, BB * NN, BC, BC, 0, 0);
  // x = x + attn_proj(attn_out)   (accumulate into F_X)
  gemm_rm(stream, W + F_AOUT, F(P_APW), F(P_APB), W + F_X, BB * NN, HH, BC, 0, 1);

  // stats head + group norm (kept in (B, 2H, N) layout == x_m/x_logs views)
  gemm_rm(stream, W + F_X, F(P_ST1W), F(P_ST1B), W + F_S1, BB * NN, HH, HH, 1, 0);
  gemm_rm(stream, W + F_S1, F(P_ST2W), F(P_ST2B), W + F_S2, BB * NN, HH, HH, 1, 0);
  gemm_rm(stream, W + F_S2, F(P_ST3W), F(P_ST3B), W + F_S, BB * NN, 2 * HH, HH, 0, 0);
  k_transpose<<<g1((long)BB * NN * 2 * HH), 256, 0, stream>>>(W + F_S, W + F_ST, NN, 2 * HH);
  k_gn_stats<<<BB * 8, 256, 0, stream>>>(W + F_ST, W + F_STATS, 2 * HH, NN, 8);
  k_gn_apply<<<g1((long)BB * 2 * HH * NN), 256, 0, stream>>>(W + F_ST, W + F_STATS, F(P_ANG), F(P_ANB), W + F_ST, 2 * HH, NN, 8, 0);

  // Gaussian log-likelihood: two batched WMMA GEMMs over K=H
  k_rT_xmr<<<g1((long)BB * NN * HH), 256, 0, stream>>>(W + F_ST, W + F_RT, W + F_XMR);
  k_mzc2<<<g1((long)BB * TT * HH), 256, 0, stream>>>(W + F_ZP, W + F_MZ);
  gemm_full(stream, W + F_RT, W + F_MZ, nullptr, W + F_LOGP, NN, TT, HH,
            HH, HH, TT, (long)NN * HH, (long)TT * HH, (long)NN * TT, BB, 0, 0);
  gemm_full(stream, W + F_XMR, W + F_ZP, nullptr, W + F_LOGP, NN, TT, HH,
            HH, HH, TT, (long)NN * HH, (long)TT * HH, (long)NN * TT, BB, 0, 1);
  k_l14<<<g1(BB * NN), 256, 0, stream>>>(W + F_ST, W + F_RT, W + F_L14);
  k_addl14<<<g1((long)BB * NN * TT), 256, 0, stream>>>(W + F_LOGP, W + F_L14);

  // monotonic alignment DP + gather
  k_mp_forward<<<BB, NN, 0, stream>>>(W + F_LOGP, xlens, zlens, W + F_QDP);
  k_mp_backtrack<<<1, 32, 0, stream>>>(W + F_QDP, xlens, zlens, (int*)(W + F_IDX));
  k_gather<<<g1((long)BB * HH * TT), 256, 0, stream>>>(W + F_ST, (const int*)(W + F_IDX), outAM, outAL);
  k_sampled<<<g1((long)BB * TT * HH), 256, 0, stream>>>(outAM, outAL, F(IN_NOISE), zlens, W + F_SAMP);

  // mamba decoder (run twice, shared weights)
  auto mamba = [&](const float* min_, float* mout) {
    float* hbuf = W + F_HBUF;  float* mtmp = W + F_MTMP; float* xz = W + F_XZ;
    float* xconv = W + F_XCONV; float* dbc = W + F_DBC;  float* delta = W + F_DELTA;
    float* ys = W + F_YS;       float* x2 = W + F_X2;    float* ffh = W + F_FFH;
    k_ln<<<BB * TT, 256, 0, stream>>>(min_, F(P_LN1G), F(P_LN1B), hbuf, HH, 0);
    for (int ly = 0; ly < 2; ++ly) {
      int pb = ly ? P_LAYER1 : P_LAYER0;
      k_ln<<<BB * TT, 256, 0, stream>>>(hbuf, F(pb + 0), nullptr, mtmp, HH, 1);
      gemm_rm(stream, mtmp, F(pb + 1), nullptr, xz, BB * TT, 2 * D_INNER, HH, 0, 0);
      k_conv_dw<<<g1((long)BB * TT * D_INNER), 256, 0, stream>>>(xz, F(pb + 2), F(pb + 3), xconv, TT);
      gemm_rm(stream, xconv, F(pb + 4), nullptr, dbc, BB * TT, 64, D_INNER, 0, 0);
      gemm_full(stream, dbc, F(pb + 5), F(pb + 6), delta, BB * TT, D_INNER, DT_RANK,
                64, DT_RANK, D_INNER, 0, 0, 0, 1, 3 /*softplus*/, 0);
      k_scan<<<g1(BB * D_INNER), 256, 0, stream>>>(delta, dbc, xconv, F(pb + 7), F(pb + 8), ys, TT);
      k_resmul<<<g1((long)BB * TT * D_INNER), 256, 0, stream>>>(ys, xz, TT);
      gemm_rm(stream, ys, F(pb + 9), nullptr, hbuf, BB * TT, HH, D_INNER, 0, 1 /*acc*/);
    }
    k_add<<<g1((long)BB * TT * HH), 256, 0, stream>>>(min_, hbuf, x2, (long)BB * TT * HH);
    k_ln<<<BB * TT, 256, 0, stream>>>(x2, F(P_LN2G), F(P_LN2B), mtmp, HH, 0);
    gemm_rm(stream, mtmp, F(P_FFN1W), F(P_FFN1B), ffh, BB * TT, 2 * HH, HH, 2 /*relu*/, 0);
    gemm_rm(stream, ffh, F(P_FFN2W), F(P_FFN2B), x2, BB * TT, HH, 2 * HH, 0, 1 /*acc*/);
    gemm_rm(stream, x2, F(P_FPW), F(P_FPB), mout, BB * TT, SP_C, HH, 0, 0);
  };
  mamba(outLAT, out + OUT_TRUE);
  mamba(W + F_SAMP, out + OUT_PRED);
}
